// MultiHeadSelfAttention_16698832847294
// MI455X (gfx1250) — matrix operand order repaired and checked by reading, not compiled, not run on hardware
//
#include <hip/hip_runtime.h>
#include <hip/hip_bf16.h>

typedef __attribute__((ext_vector_type(16))) _Float16 v16h;
typedef __attribute__((ext_vector_type(8)))  _Float16 v8h;
typedef __attribute__((ext_vector_type(8)))  float    v8f;

static constexpr int B_  = 2;
static constexpr int S_  = 2048;
static constexpr int D_  = 1024;
static constexpr int H_  = 16;
static constexpr int DK_ = 64;
static constexpr int M_  = B_ * S_;   // 4096 flattened (b,s) rows

// ---------------- fragment helpers (wave32 WMMA f16 layouts) ----------------

__device__ __forceinline__ v8f zero8() { v8f z = {}; return z; }

__device__ __forceinline__ v16h mk16(v8h lo, v8h hi) {
  return __builtin_shufflevector(lo, hi, 0,1,2,3,4,5,6,7,8,9,10,11,12,13,14,15);
}

// A matrix 16x32 (MxK) fragment from row-major [*, ld] f16.
__device__ __forceinline__ v16h load_a_frag(const _Float16* __restrict__ base,
                                            int ld, int mr, int kc, int lane) {
  const int m  = mr + (lane & 15);
  const int kh = lane >> 4;
  const _Float16* p = base + (size_t)m * ld + kc + kh * 8;
  return mk16(*(const v8h*)p, *(const v8h*)(p + 16));
}

// B matrix 32x16 (KxN) fragment from rows nbase..nbase+15 of a row-major
// [*, ld] matrix (computes the X @ W^T pattern).
__device__ __forceinline__ v16h load_b_frag(const _Float16* __restrict__ base,
                                            int ld, int nbase, int kc, int lane) {
  const int n  = nbase + (lane & 15);
  const int kh = lane >> 4;
  const _Float16* p = base + (size_t)n * ld + kc + kh * 8;
  return mk16(*(const v8h*)p, *(const v8h*)(p + 16));
}

__device__ __forceinline__ v8f wmma_f16(v16h a, v16h b, v8f c) {
  return __builtin_amdgcn_wmma_f32_16x16x32_f16(false, a, false, b,
                                                (short)0, c, false, false);
}

__device__ __forceinline__ float rmax16(float v) {
  v = fmaxf(v, __shfl_xor(v, 1));
  v = fmaxf(v, __shfl_xor(v, 2));
  v = fmaxf(v, __shfl_xor(v, 4));
  v = fmaxf(v, __shfl_xor(v, 8));
  return v;
}
__device__ __forceinline__ float rsum16(float v) {
  v += __shfl_xor(v, 1);
  v += __shfl_xor(v, 2);
  v += __shfl_xor(v, 4);
  v += __shfl_xor(v, 8);
  return v;
}

// ---------------------------- kernels ---------------------------------------

__global__ __launch_bounds__(256) void cvt_kernel(const float* __restrict__ s,
                                                  _Float16* __restrict__ d,
                                                  int n) {
  for (int i = blockIdx.x * blockDim.x + threadIdx.x; i < n;
       i += gridDim.x * blockDim.x)
    d[i] = (_Float16)s[i];
}

// ---------------- fused QKV projection (double-buffered) --------------------

struct QkvFrags { v16h a, q0, q1, k0, k1, v0, v1; };

__device__ __forceinline__ QkvFrags qkv_load(const _Float16* __restrict__ X,
                                             const _Float16* __restrict__ Wq,
                                             const _Float16* __restrict__ Wk,
                                             const _Float16* __restrict__ Wv,
                                             int mr, int nr, int kc, int lane) {
  QkvFrags f;
  f.a  = load_a_frag(X,  D_, mr,      kc, lane);
  f.q0 = load_b_frag(Wq, D_, nr,      kc, lane);
  f.q1 = load_b_frag(Wq, D_, nr + 16, kc, lane);
  f.k0 = load_b_frag(Wk, D_, nr,      kc, lane);
  f.k1 = load_b_frag(Wk, D_, nr + 16, kc, lane);
  f.v0 = load_b_frag(Wv, D_, nr,      kc, lane);
  f.v1 = load_b_frag(Wv, D_, nr + 16, kc, lane);
  return f;
}

__global__ __launch_bounds__(256) void qkv_proj_kernel(
    const _Float16* __restrict__ X,                         // [M_, D_]
    const _Float16* __restrict__ Wq,
    const _Float16* __restrict__ Wk,
    const _Float16* __restrict__ Wv,                        // each [D_, D_]
    _Float16* __restrict__ Qh,                              // [B,H,S,DK]
    _Float16* __restrict__ Kh,                              // [B,H,S,DK]
    _Float16* __restrict__ Vt)                              // [B,H,DK,S]
{
  const int lane = threadIdx.x & 31;
  const int wv   = threadIdx.x >> 5;
  const int nr   = blockIdx.x * 32;
  const int mr   = blockIdx.y * 128 + wv * 16;

  v8f aq0 = zero8(), aq1 = zero8();
  v8f ak0 = zero8(), ak1 = zero8();
  v8f av0 = zero8(), av1 = zero8();

  QkvFrags cur = qkv_load(X, Wq, Wk, Wv, mr, nr, 0, lane);
  for (int kc = 32; kc < D_; kc += 32) {
    QkvFrags nxt = qkv_load(X, Wq, Wk, Wv, mr, nr, kc, lane);  // prefetch
    aq0 = wmma_f16(cur.a, cur.q0, aq0); aq1 = wmma_f16(cur.a, cur.q1, aq1);
    ak0 = wmma_f16(cur.a, cur.k0, ak0); ak1 = wmma_f16(cur.a, cur.k1, ak1);
    av0 = wmma_f16(cur.a, cur.v0, av0); av1 = wmma_f16(cur.a, cur.v1, av1);
    cur = nxt;
  }
  aq0 = wmma_f16(cur.a, cur.q0, aq0); aq1 = wmma_f16(cur.a, cur.q1, aq1);
  ak0 = wmma_f16(cur.a, cur.k0, ak0); ak1 = wmma_f16(cur.a, cur.k1, ak1);
  av0 = wmma_f16(cur.a, cur.v0, av0); av1 = wmma_f16(cur.a, cur.v1, av1);

  const int kh  = lane >> 4;
  const int col = lane & 15;
#pragma unroll
  for (int nn = 0; nn < 2; ++nn) {
    const int cg = nr + nn * 16 + col;   // global output column (head-major)
    const int h  = cg >> 6;
    const int dk = cg & 63;
#pragma unroll
    for (int j = 0; j < 8; ++j) {
      const int rg = mr + j + 8 * kh;    // global row = b*S + s
      const int b  = rg >> 11;
      const int s  = rg & 2047;
      const size_t bh = (size_t)(b * H_ + h);
      Qh[(bh * S_ + s) * DK_ + dk] = (_Float16)(nn ? aq1[j] : aq0[j]);
      Kh[(bh * S_ + s) * DK_ + dk] = (_Float16)(nn ? ak1[j] : ak0[j]);
      Vt[(bh * DK_ + dk) * S_ + s] = (_Float16)(nn ? av1[j] : av0[j]);
    }
  }
}

// ---------------- single-pass causal flash attention ------------------------
// One 16-row q tile per wave, 32 keys/iteration.  V loads for the current
// tile and K loads for the next tile are issued before the softmax block so
// their latency hides under VALU/shuffle/LDS work.  No block barriers.

__global__ __launch_bounds__(256) void flash_attn_kernel(
    const _Float16* __restrict__ Qh, const _Float16* __restrict__ Kh,
    const _Float16* __restrict__ Vt, _Float16* __restrict__ Oh)
{
  __shared__ _Float16 Pt[8][16 * 32];   // per-wave 16x32 P tile (8 KB total)

  const int lane = threadIdx.x & 31;
  const int wv   = threadIdx.x >> 5;
  const int kh   = lane >> 4;
  const int col  = lane & 15;
  const int b    = blockIdx.z;
  const int h    = blockIdx.y;
  const int qr   = blockIdx.x * 128 + wv * 16;

  const size_t bh = (size_t)(b * H_ + h);
  const _Float16* Qb = Qh + bh * (size_t)S_ * DK_;
  const _Float16* Kb = Kh + bh * (size_t)S_ * DK_;
  const _Float16* Vb = Vt + bh * (size_t)DK_ * S_;

  const v16h qa0 = load_a_frag(Qb, DK_, qr, 0,  lane);
  const v16h qa1 = load_a_frag(Qb, DK_, qr, 32, lane);

  v8f o[4];
#pragma unroll
  for (int i = 0; i < 4; ++i) o[i] = zero8();
  float m[8], l[8];
#pragma unroll
  for (int j = 0; j < 8; ++j) { m[j] = -1.0e30f; l[j] = 0.0f; }

  // base-2 softmax: c2 = log2(e)/sqrt(dk); exp2((s-m)*c2) == exp((s-m)/sqrt(dk))
  const float c2   = 0.18033688011112042f;
  const int   kend = qr + 16;           // causal: keys < qr+16

  v16h kf0 = load_b_frag(Kb, DK_, 0,  0,  lane);
  v16h kf1 = load_b_frag(Kb, DK_, 0,  32, lane);
  v16h kf2 = load_b_frag(Kb, DK_, 16, 0,  lane);
  v16h kf3 = load_b_frag(Kb, DK_, 16, 32, lane);

  for (int kb = 0; kb < kend; kb += 32) {
    // scores for this key tile
    v8f s0 = zero8(), s1 = zero8();
    s0 = wmma_f16(qa0, kf0, s0);
    s0 = wmma_f16(qa1, kf1, s0);
    s1 = wmma_f16(qa0, kf2, s1);
    s1 = wmma_f16(qa1, kf3, s1);

    // issue V loads (consumed after softmax) and next-tile K loads now
    v16h vf0 = load_b_frag(Vb, S_, 0,  kb, lane);
    v16h vf1 = load_b_frag(Vb, S_, 16, kb, lane);
    v16h vf2 = load_b_frag(Vb, S_, 32, kb, lane);
    v16h vf3 = load_b_frag(Vb, S_, 48, kb, lane);
    const int  kb2  = kb + 32;
    const bool more = kb2 < kend;   // wave-uniform
    if (more) {
      kf0 = load_b_frag(Kb, DK_, kb2,      0,  lane);
      kf1 = load_b_frag(Kb, DK_, kb2,      32, lane);
      kf2 = load_b_frag(Kb, DK_, kb2 + 16, 0,  lane);
      kf3 = load_b_frag(Kb, DK_, kb2 + 16, 32, lane);
    }

    // online softmax (base-2 domain)
    float p0[8], p1[8], alpha[8];
#pragma unroll
    for (int j = 0; j < 8; ++j) {
      const int q = qr + j + 8 * kh;
      float v0 = (kb + col      <= q) ? s0[j] * c2 : -1.0e30f;
      float v1 = (kb + 16 + col <= q) ? s1[j] * c2 : -1.0e30f;
      float mt = rmax16(fmaxf(v0, v1));
      float mn = fmaxf(m[j], mt);
      alpha[j] = exp2f(m[j] - mn);
      p0[j] = exp2f(v0 - mn);
      p1[j] = exp2f(v1 - mn);
      l[j]  = l[j] * alpha[j] + rsum16(p0[j] + p1[j]);
      m[j]  = mn;
    }

    // stash P (C/D layout) to LDS, reload in A-fragment layout
#pragma unroll
    for (int j = 0; j < 8; ++j) {
      const int r = j + 8 * kh;
      Pt[wv][r * 32 + col]      = (_Float16)p0[j];
      Pt[wv][r * 32 + 16 + col] = (_Float16)p1[j];
    }

    // rescale O while the DS stores are in flight
#pragma unroll
    for (int i = 0; i < 4; ++i)
#pragma unroll
      for (int j = 0; j < 8; ++j)
        o[i][j] *= alpha[j];

    const _Float16* lp = &Pt[wv][col * 32 + kh * 8];
    const v16h pa = mk16(*(const v8h*)lp, *(const v8h*)(lp + 16));

    o[0] = wmma_f16(pa, vf0, o[0]);
    o[1] = wmma_f16(pa, vf1, o[1]);
    o[2] = wmma_f16(pa, vf2, o[2]);
    o[3] = wmma_f16(pa, vf3, o[3]);
  }

  float inv[8];
#pragma unroll
  for (int j = 0; j < 8; ++j) inv[j] = 1.0f / l[j];
#pragma unroll
  for (int i = 0; i < 4; ++i)
#pragma unroll
    for (int j = 0; j < 8; ++j) {
      const int s = qr + j + 8 * kh;
      Oh[((size_t)(b * S_ + s)) * D_ + h * DK_ + i * 16 + col] =
          (_Float16)(o[i][j] * inv[j]);
    }
}

// ---------------- output projection (double-buffered) -----------------------

struct OutFrags { v16h a0, a1, a2, a3, b0, b1; };

__device__ __forceinline__ OutFrags out_load(const _Float16* __restrict__ A,
                                             const _Float16* __restrict__ Wo,
                                             int mr, int nr, int kc, int lane) {
  OutFrags f;
  f.a0 = load_a_frag(A, D_, mr,      kc, lane);
  f.a1 = load_a_frag(A, D_, mr + 16, kc, lane);
  f.a2 = load_a_frag(A, D_, mr + 32, kc, lane);
  f.a3 = load_a_frag(A, D_, mr + 48, kc, lane);
  f.b0 = load_b_frag(Wo, D_, nr,      kc, lane);
  f.b1 = load_b_frag(Wo, D_, nr + 16, kc, lane);
  return f;
}

__global__ __launch_bounds__(256) void out_proj_kernel(
    const _Float16* __restrict__ A,    // [M_, D_] attention output
    const _Float16* __restrict__ Wo,   // [D_, D_]
    float* __restrict__ C)             // [M_, D_]
{
  const int lane = threadIdx.x & 31;
  const int wv   = threadIdx.x >> 5;
  const int mr   = blockIdx.y * 128 + (wv & 1) * 64;
  const int nr   = blockIdx.x * 128 + (wv >> 1) * 32;

  v8f acc[4][2];
#pragma unroll
  for (int i = 0; i < 4; ++i) { acc[i][0] = zero8(); acc[i][1] = zero8(); }

  OutFrags cur = out_load(A, Wo, mr, nr, 0, lane);
  for (int kc = 32; kc < D_; kc += 32) {
    OutFrags nxt = out_load(A, Wo, mr, nr, kc, lane);  // prefetch
    acc[0][0] = wmma_f16(cur.a0, cur.b0, acc[0][0]);
    acc[0][1] = wmma_f16(cur.a0, cur.b1, acc[0][1]);
    acc[1][0] = wmma_f16(cur.a1, cur.b0, acc[1][0]);
    acc[1][1] = wmma_f16(cur.a1, cur.b1, acc[1][1]);
    acc[2][0] = wmma_f16(cur.a2, cur.b0, acc[2][0]);
    acc[2][1] = wmma_f16(cur.a2, cur.b1, acc[2][1]);
    acc[3][0] = wmma_f16(cur.a3, cur.b0, acc[3][0]);
    acc[3][1] = wmma_f16(cur.a3, cur.b1, acc[3][1]);
    cur = nxt;
  }
  acc[0][0] = wmma_f16(cur.a0, cur.b0, acc[0][0]);
  acc[0][1] = wmma_f16(cur.a0, cur.b1, acc[0][1]);
  acc[1][0] = wmma_f16(cur.a1, cur.b0, acc[1][0]);
  acc[1][1] = wmma_f16(cur.a1, cur.b1, acc[1][1]);
  acc[2][0] = wmma_f16(cur.a2, cur.b0, acc[2][0]);
  acc[2][1] = wmma_f16(cur.a2, cur.b1, acc[2][1]);
  acc[3][0] = wmma_f16(cur.a3, cur.b0, acc[3][0]);
  acc[3][1] = wmma_f16(cur.a3, cur.b1, acc[3][1]);

  const int kh = lane >> 4, col = lane & 15;
#pragma unroll
  for (int i = 0; i < 4; ++i)
#pragma unroll
    for (int nn = 0; nn < 2; ++nn)
#pragma unroll
      for (int j = 0; j < 8; ++j)
        C[(size_t)(mr + 16 * i + j + 8 * kh) * D_ + nr + 16 * nn + col] =
            acc[i][nn][j];
}

// ---------------------------- launcher ---------------------------------------

extern "C" void kernel_launch(void* const* d_in, const int* in_sizes, int n_in,
                              void* d_out, int out_size, void* d_ws, size_t ws_size,
                              hipStream_t stream) {
  (void)in_sizes; (void)n_in; (void)out_size; (void)ws_size;

  const float* X  = (const float*)d_in[0];
  const float* Wq = (const float*)d_in[1];
  const float* Wk = (const float*)d_in[2];
  const float* Wv = (const float*)d_in[3];
  const float* Wo = (const float*)d_in[4];

  const size_t NX = (size_t)M_ * D_;   // 4,194,304 elements
  const size_t NW = (size_t)D_ * D_;   // 1,048,576 elements

  // Workspace layout (f16): X | Wq | Wk | Wv | Wo | Q | K | Vt  = 40 MB total.
  _Float16* Xh  = (_Float16*)d_ws;
  _Float16* Wqh = Xh  + NX;
  _Float16* Wkh = Wqh + NW;
  _Float16* Wvh = Wkh + NW;
  _Float16* Woh = Wvh + NW;
  _Float16* Qh  = Woh + NW;
  _Float16* Kh  = Qh  + NX;
  _Float16* Vt  = Kh  + NX;

  cvt_kernel<<<2048, 256, 0, stream>>>(X,  Xh,  (int)NX);
  cvt_kernel<<<1024, 256, 0, stream>>>(Wq, Wqh, (int)NW);
  cvt_kernel<<<1024, 256, 0, stream>>>(Wk, Wkh, (int)NW);
  cvt_kernel<<<1024, 256, 0, stream>>>(Wv, Wvh, (int)NW);
  cvt_kernel<<<1024, 256, 0, stream>>>(Wo, Woh, (int)NW);

  qkv_proj_kernel<<<dim3(D_ / 32, M_ / 128), 256, 0, stream>>>(
      Xh, Wqh, Wkh, Wvh, Qh, Kh, Vt);

  _Float16* Oh = Xh;  // alias: X is dead after the QKV projection
  flash_attn_kernel<<<dim3(S_ / 128, H_, B_), 256, 0, stream>>>(Qh, Kh, Vt, Oh);

  out_proj_kernel<<<dim3(D_ / 128, M_ / 128), 256, 0, stream>>>(
      Oh, Woh, (float*)d_out);
}
